// IntraEnvAggregator_2499670966884
// MI455X (gfx1250) — matrix operand order repaired and checked by reading, not compiled, not run on hardware
//
#include <hip/hip_runtime.h>
#include <hip/hip_fp16.h>

typedef __attribute__((ext_vector_type(16))) _Float16 v16h;
typedef __attribute__((ext_vector_type(8)))  _Float16 v8h;
typedef __attribute__((ext_vector_type(4)))  _Float16 v4h;
typedef __attribute__((ext_vector_type(8)))  float    v8f;
typedef __attribute__((ext_vector_type(4)))  float    v4f;
typedef __attribute__((ext_vector_type(4)))  unsigned int v4u;
typedef __attribute__((ext_vector_type(8)))  int      v8i;
typedef __attribute__((ext_vector_type(4)))  int      v4i;

#define DD   256
#define KK   16
#define HH   8
#define HDIM 32
#define INV_SCALE 0.17677669529663687f   // 1/sqrt(32)

union V16 { v16h v; struct { v8h lo, hi; } s; };

// ---------------------------------------------------------------------------
// Kernel 0: h (f32) -> h16 (f16), row-major, vectorized
// ---------------------------------------------------------------------------
__global__ void k_h2f16(const float* __restrict__ h, _Float16* __restrict__ h16, int total4) {
    int gid = blockIdx.x * blockDim.x + threadIdx.x;
    if (gid >= total4) return;
    v4f x = *reinterpret_cast<const v4f*>(h + (size_t)gid * 4);
    v4h y;
    y[0] = (_Float16)x[0]; y[1] = (_Float16)x[1];
    y[2] = (_Float16)x[2]; y[3] = (_Float16)x[3];
    *reinterpret_cast<v4h*>(h16 + (size_t)gid * 4) = y;
}

// ---------------------------------------------------------------------------
// Kernel 1: transpose+convert Wq, Wk (D x D) and Wo (D x O) to f16, T[c][k]=W[k][c]
// ---------------------------------------------------------------------------
__global__ void k_wt(const float* __restrict__ Wq, const float* __restrict__ Wk,
                     const float* __restrict__ Wo,
                     _Float16* __restrict__ WqT, _Float16* __restrict__ WkT,
                     _Float16* __restrict__ WoT) {
    int idx = blockIdx.x * 256 + threadIdx.x;      // 3 * 65536 total
    int m = idx >> 16;
    int r = (idx >> 8) & 255;                      // output row = column of W
    int c = idx & 255;                             // output col = row of W
    const float* W = (m == 0) ? Wq : (m == 1) ? Wk : Wo;
    _Float16*    T = (m == 0) ? WqT : (m == 1) ? WkT : WoT;
    T[r * DD + c] = (_Float16)W[c * DD + r];
}

// ---------------------------------------------------------------------------
// Kernel 2: Q = h@Wq + bq, Kall = h@Wk + bk   (grid.y selects which)
// Block: 256 thr = 8 waves; tile 16 rows x 256 cols; wave -> 2 col-tiles.
// ---------------------------------------------------------------------------
__global__ __launch_bounds__(256) void k_proj(
    const _Float16* __restrict__ h16,
    const _Float16* __restrict__ WqT, const _Float16* __restrict__ WkT,
    const float* __restrict__ bq, const float* __restrict__ bk,
    _Float16* __restrict__ q16, _Float16* __restrict__ k16, int N) {
    const _Float16* WT   = blockIdx.y ? WkT : WqT;
    const float*    bias = blockIdx.y ? bk  : bq;
    _Float16*       out  = blockIdx.y ? k16 : q16;

    int wave = threadIdx.x >> 5, lane = threadIdx.x & 31;
    int laneLow = lane & 15, laneHi = lane >> 4;
    int m0 = blockIdx.x * 16;
    int n0 = wave * 32;
    int rowA = m0 + laneLow; if (rowA >= N) rowA = N - 1;

    v8f c0 = {}, c1 = {};
    const _Float16* aBase  = h16 + (size_t)rowA * DD + laneHi * 16;
    const _Float16* b0Base = WT + (size_t)(n0 + laneLow) * DD + laneHi * 16;
    const _Float16* b1Base = WT + (size_t)(n0 + 16 + laneLow) * DD + laneHi * 16;

#pragma unroll
    for (int kb = 0; kb < 8; ++kb) {
        int kc = kb * 32;
        V16 a, b0, b1;
        a.s.lo  = *reinterpret_cast<const v8h*>(aBase + kc);
        a.s.hi  = *reinterpret_cast<const v8h*>(aBase + kc + 8);
        b0.s.lo = *reinterpret_cast<const v8h*>(b0Base + kc);
        b0.s.hi = *reinterpret_cast<const v8h*>(b0Base + kc + 8);
        b1.s.lo = *reinterpret_cast<const v8h*>(b1Base + kc);
        b1.s.hi = *reinterpret_cast<const v8h*>(b1Base + kc + 8);
        c0 = __builtin_amdgcn_wmma_f32_16x16x32_f16(false, a.v, false, b0.v, (short)0, c0, false, false);
        c1 = __builtin_amdgcn_wmma_f32_16x16x32_f16(false, a.v, false, b1.v, (short)0, c1, false, false);
    }

    float bias0 = bias[n0 + laneLow];
    float bias1 = bias[n0 + 16 + laneLow];
#pragma unroll
    for (int r = 0; r < 8; ++r) {
        int row = m0 + r + laneHi * 8;
        if (row < N) {
            out[(size_t)row * DD + n0 + laneLow]      = (_Float16)(c0[r] + bias0);
            out[(size_t)row * DD + n0 + 16 + laneLow] = (_Float16)(c1[r] + bias1);
        }
    }
}

// ---------------------------------------------------------------------------
// Kernel 3: compaction + sigmoid->softmax attention + ctx (+residual) -> f16
// Block: 256 thr = 8 waves, one node per wave.
// ---------------------------------------------------------------------------
__global__ __launch_bounds__(256) void k_attn(
    const float* __restrict__ h, const int* __restrict__ nbr_idx,
    const unsigned char* __restrict__ nbr_mask,
    const _Float16* __restrict__ q16, const _Float16* __restrict__ k16,
    const float* __restrict__ bk, _Float16* __restrict__ ctx16, int N) {
    __shared__ int s_idx[8][KK];
    int wave = threadIdx.x >> 5, lane = threadIdx.x & 31;
    int node = blockIdx.x * 8 + wave;
    bool active = node < N;
    int nn = active ? node : 0;

    // --- stable compaction: valid neighbors to front ---
    bool m = false; int myIdx = 0;
    if (lane < KK) {
        m = nbr_mask[(size_t)nn * KK + lane] != 0;
        myIdx = nbr_idx[(size_t)nn * KK + lane];
    }
    unsigned long long bal = __ballot(m);
    unsigned bits = (unsigned)bal & 0xFFFFu;
    int count = __popc(bits);
    if (lane < KK) {
        unsigned below = (1u << lane) - 1u;
        int dest = m ? __popc(bits & below)
                     : count + __popc((~bits) & 0xFFFFu & below);
        s_idx[wave][dest] = myIdx;
    }
    __syncthreads();
    if (!active) return;

    int hh = lane >> 2, g = lane & 3;

    float qf[HDIM];
    const _Float16* qp = q16 + (size_t)node * DD + hh * HDIM;
#pragma unroll
    for (int e = 0; e < HDIM; ++e) qf[e] = (float)qp[e];

    float sg[4];
#pragma unroll
    for (int t = 0; t < 4; ++t) {
        int j = g * 4 + t;
        float s = 0.f;
        if (j < count) {
            const _Float16* kp = k16 + (size_t)s_idx[wave][j] * DD + hh * HDIM;
#pragma unroll
            for (int e = 0; e < HDIM; ++e) s += qf[e] * (float)kp[e];
        } else {
            const float* bp = bk + hh * HDIM;   // zero-row @ Wk + b = bias
#pragma unroll
            for (int e = 0; e < HDIM; ++e) s += qf[e] * bp[e];
        }
        s *= INV_SCALE;
        bool keep = (bits >> j) & 1u;
        sg[t] = keep ? (1.f / (1.f + __expf(-s))) : 0.f;  // sigmoid(-inf)=0
    }

    float ex[4], esum = 0.f;
#pragma unroll
    for (int t = 0; t < 4; ++t) { ex[t] = __expf(sg[t]); esum += ex[t]; }
    esum += __shfl_xor(esum, 1);
    esum += __shfl_xor(esum, 2);
    float inv = 1.f / esum;
    float at[4];
#pragma unroll
    for (int t = 0; t < 4; ++t) at[t] = ex[t] * inv;

    float a16[16];
    int base = lane & ~3;
#pragma unroll
    for (int g2 = 0; g2 < 4; ++g2) {
        a16[g2 * 4 + 0] = __shfl(at[0], base + g2);
        a16[g2 * 4 + 1] = __shfl(at[1], base + g2);
        a16[g2 * 4 + 2] = __shfl(at[2], base + g2);
        a16[g2 * 4 + 3] = __shfl(at[3], base + g2);
    }

    int d0 = lane * 8;
    float acc[8] = {0.f, 0.f, 0.f, 0.f, 0.f, 0.f, 0.f, 0.f};
#pragma unroll
    for (int j = 0; j < KK; ++j) {
        float a = a16[j];
        if (j < count) {
            v8h kv = *reinterpret_cast<const v8h*>(k16 + (size_t)s_idx[wave][j] * DD + d0);
#pragma unroll
            for (int e = 0; e < 8; ++e) acc[e] += a * (float)kv[e];
        } else {
#pragma unroll
            for (int e = 0; e < 8; ++e) acc[e] += a * bk[d0 + e];
        }
    }
    const float* hp = h + (size_t)node * DD + d0;
    v8h o;
#pragma unroll
    for (int e = 0; e < 8; ++e) o[e] = (_Float16)(acc[e] + hp[e]);
    *reinterpret_cast<v8h*>(ctx16 + (size_t)node * DD + d0) = o;
}

// ---------------------------------------------------------------------------
// TDM helper: async 2D tile load Global -> LDS (f16 elements).
// Builds the D# per CDNA5 ISA ch.8: group0 {count, lds_addr, global_addr, type=2},
// group1 {data_size=2B, tensor dims, tile dims, dim0 stride}. Groups 2/3 zero
// (tile_dim2..4 = 0 -> unused). Tracked by TENSORcnt.
// ---------------------------------------------------------------------------
#if __has_builtin(__builtin_amdgcn_tensor_load_to_lds)
#define USE_TDM 1
__device__ __forceinline__ void tdm_load_2d_f16(
    unsigned lds_byte_addr, const void* gaddr,
    unsigned tile_k, unsigned tile_n,
    unsigned tensor_d0, unsigned tensor_d1, unsigned stride0) {
    unsigned long long ga = (unsigned long long)gaddr;
    v4u g0;
    g0[0] = 1u;                                            // count=1 (user mode)
    g0[1] = lds_byte_addr;                                 // lds_addr
    g0[2] = (unsigned)(ga & 0xFFFFFFFFu);                  // global_addr[31:0]
    g0[3] = (unsigned)((ga >> 32) & 0x01FFFFFFu)           // global_addr[56:32]
            | (2u << 30);                                  // type=2 ("image")
    v8i g1;
    g1[0] = (int)(1u << 16);                               // data_size=1 -> 2 bytes
    g1[1] = (int)((tensor_d0 & 0xFFFFu) << 16);            // tensor_dim0[15:0]
    g1[2] = (int)(((tensor_d0 >> 16) & 0xFFFFu)
                  | ((tensor_d1 & 0xFFFFu) << 16));        // td0[31:16] | td1[15:0]
    g1[3] = (int)(((tensor_d1 >> 16) & 0xFFFFu)
                  | ((tile_k & 0xFFFFu) << 16));           // td1[31:16] | tile_dim0
    g1[4] = (int)(tile_n & 0xFFFFu);                       // tile_dim1; tile_dim2=0
    g1[5] = (int)stride0;                                  // tensor_dim0_stride[31:0]
    g1[6] = 0;                                             // stride0[47:32]; ts1[15:0]
    g1[7] = 0;
    v4i g2; g2[0] = 0; g2[1] = 0; g2[2] = 0; g2[3] = 0;
    v4i g3; g3[0] = 0; g3[1] = 0; g3[2] = 0; g3[3] = 0;
#if defined(__clang_major__) && (__clang_major__ >= 23)
    v8i g4; g4[0] = 0; g4[1] = 0; g4[2] = 0; g4[3] = 0;
    g4[4] = 0; g4[5] = 0; g4[6] = 0; g4[7] = 0;
    __builtin_amdgcn_tensor_load_to_lds(g0, g1, g2, g3, g4, 0);
#else
    __builtin_amdgcn_tensor_load_to_lds(g0, g1, g2, g3, 0);
#endif
}
#else
#define USE_TDM 0
#endif

// ---------------------------------------------------------------------------
// Kernel 4: out = ctx@Wo + bo, then fused LayerNorm -> f32 d_out.
// B (WoT) is staged through LDS in 4 chunks of 256 rows x 64 K-elems (32KB)
// by the Tensor Data Mover; WMMAs consume B via ds_load_b128. The LDS chunk
// is unioned with the LayerNorm staging tile (weights dead by epilogue).
// ---------------------------------------------------------------------------
#define KCH 64   // K elements per staged chunk

__global__ __launch_bounds__(256) void k_out(
    const _Float16* __restrict__ ctx16, const _Float16* __restrict__ WoT,
    const float* __restrict__ bo, const float* __restrict__ gamma,
    const float* __restrict__ beta, float* __restrict__ out, int N) {
    __shared__ union {
        _Float16 wt[DD * KCH];       // 32KB: B chunk, row n -> KCH k-elems packed
        float    st[16][DD + 4];     // LayerNorm staging tile
    } sm;
    int wave = threadIdx.x >> 5, lane = threadIdx.x & 31;
    int laneLow = lane & 15, laneHi = lane >> 4;
    int m0 = blockIdx.x * 16;
    int n0 = wave * 32;
    int rowA = m0 + laneLow; if (rowA >= N) rowA = N - 1;

    v8f c0 = {}, c1 = {};
    const _Float16* aBase = ctx16 + (size_t)rowA * DD + laneHi * 16;

#if USE_TDM
    const _Float16* bl0 = sm.wt + (n0 + laneLow) * KCH + laneHi * 16;
    const _Float16* bl1 = sm.wt + (n0 + 16 + laneLow) * KCH + laneHi * 16;
#pragma unroll
    for (int ch = 0; ch < DD / KCH; ++ch) {
        if (threadIdx.x < 32) {   // wave 0 issues the DMA (EXEC ignored by TDM)
            tdm_load_2d_f16(/*lds*/0u, WoT + ch * KCH,
                            /*tile_k=*/KCH, /*tile_n=*/DD,
                            /*tensor_d0=*/DD, /*tensor_d1=*/DD, /*stride0=*/DD);
            __builtin_amdgcn_s_wait_tensorcnt(0);
        }
        __syncthreads();          // LDS chunk visible to all waves
#pragma unroll
        for (int i = 0; i < KCH / 32; ++i) {
            int kc = ch * KCH + i * 32;   // global K (A operand)
            int kl = i * 32;              // K within LDS chunk (B operand)
            V16 a, b0, b1;
            a.s.lo  = *reinterpret_cast<const v8h*>(aBase + kc);
            a.s.hi  = *reinterpret_cast<const v8h*>(aBase + kc + 8);
            b0.s.lo = *reinterpret_cast<const v8h*>(bl0 + kl);
            b0.s.hi = *reinterpret_cast<const v8h*>(bl0 + kl + 8);
            b1.s.lo = *reinterpret_cast<const v8h*>(bl1 + kl);
            b1.s.hi = *reinterpret_cast<const v8h*>(bl1 + kl + 8);
            c0 = __builtin_amdgcn_wmma_f32_16x16x32_f16(false, a.v, false, b0.v, (short)0, c0, false, false);
            c1 = __builtin_amdgcn_wmma_f32_16x16x32_f16(false, a.v, false, b1.v, (short)0, c1, false, false);
        }
        __syncthreads();          // chunk fully consumed before next DMA overwrites
    }
#else
    const _Float16* b0Base = WoT + (size_t)(n0 + laneLow) * DD + laneHi * 16;
    const _Float16* b1Base = WoT + (size_t)(n0 + 16 + laneLow) * DD + laneHi * 16;
#pragma unroll
    for (int kb = 0; kb < 8; ++kb) {
        int kc = kb * 32;
        V16 a, b0, b1;
        a.s.lo  = *reinterpret_cast<const v8h*>(aBase + kc);
        a.s.hi  = *reinterpret_cast<const v8h*>(aBase + kc + 8);
        b0.s.lo = *reinterpret_cast<const v8h*>(b0Base + kc);
        b0.s.hi = *reinterpret_cast<const v8h*>(b0Base + kc + 8);
        b1.s.lo = *reinterpret_cast<const v8h*>(b1Base + kc);
        b1.s.hi = *reinterpret_cast<const v8h*>(b1Base + kc + 8);
        c0 = __builtin_amdgcn_wmma_f32_16x16x32_f16(false, a.v, false, b0.v, (short)0, c0, false, false);
        c1 = __builtin_amdgcn_wmma_f32_16x16x32_f16(false, a.v, false, b1.v, (short)0, c1, false, false);
    }
    __syncthreads();
#endif

    float bias0 = bo[n0 + laneLow];
    float bias1 = bo[n0 + 16 + laneLow];
#pragma unroll
    for (int r = 0; r < 8; ++r) {
        sm.st[r + laneHi * 8][n0 + laneLow]      = c0[r] + bias0;
        sm.st[r + laneHi * 8][n0 + 16 + laneLow] = c1[r] + bias1;
    }
    __syncthreads();

    // LayerNorm: each wave handles 2 rows; 32 lanes x 8 elems each
#pragma unroll
    for (int rr = 0; rr < 2; ++rr) {
        int row = wave * 2 + rr;
        float x[8], sum = 0.f, sq = 0.f;
#pragma unroll
        for (int e = 0; e < 8; ++e) {
            x[e] = sm.st[row][lane * 8 + e];
            sum += x[e]; sq += x[e] * x[e];
        }
#pragma unroll
        for (int off = 16; off >= 1; off >>= 1) {
            sum += __shfl_xor(sum, off);
            sq  += __shfl_xor(sq, off);
        }
        float mu  = sum * (1.f / DD);
        float var = sq * (1.f / DD) - mu * mu;
        float rs  = rsqrtf(var + 1e-5f);
        int grow = m0 + row;
        if (grow < N) {
            float* op = out + (size_t)grow * DD + lane * 8;
#pragma unroll
            for (int e = 0; e < 8; ++e)
                op[e] = (x[e] - mu) * rs * gamma[lane * 8 + e] + beta[lane * 8 + e];
        }
    }
}

// ---------------------------------------------------------------------------
extern "C" void kernel_launch(void* const* d_in, const int* in_sizes, int n_in,
                              void* d_out, int out_size, void* d_ws, size_t ws_size,
                              hipStream_t stream) {
    const float*         h        = (const float*)d_in[0];
    const int*           nbr_idx  = (const int*)d_in[1];
    const unsigned char* nbr_mask = (const unsigned char*)d_in[2];
    const float*         Wq_w     = (const float*)d_in[3];
    const float*         Wq_b     = (const float*)d_in[4];
    const float*         Wk_w     = (const float*)d_in[5];
    const float*         Wk_b     = (const float*)d_in[6];
    const float*         Wo_w     = (const float*)d_in[7];
    const float*         Wo_b     = (const float*)d_in[8];
    const float*         gamma    = (const float*)d_in[9];
    const float*         beta     = (const float*)d_in[10];
    float*               out      = (float*)d_out;

    const int N = in_sizes[0] / DD;

    auto align256 = [](size_t x) { return (x + 255) & ~(size_t)255; };
    size_t rowbuf = align256((size_t)N * DD * sizeof(_Float16));
    size_t wtbuf  = align256((size_t)DD * DD * sizeof(_Float16));
    char* base = (char*)d_ws;
    _Float16* h16   = (_Float16*)(base);
    _Float16* q16   = (_Float16*)(base + rowbuf);
    _Float16* k16   = (_Float16*)(base + 2 * rowbuf);
    _Float16* ctx16 = (_Float16*)(base + 3 * rowbuf);
    _Float16* WqT   = (_Float16*)(base + 4 * rowbuf);
    _Float16* WkT   = (_Float16*)(base + 4 * rowbuf + wtbuf);
    _Float16* WoT   = (_Float16*)(base + 4 * rowbuf + 2 * wtbuf);

    int total4 = (N * DD) / 4;
    k_h2f16<<<(total4 + 255) / 256, 256, 0, stream>>>(h, h16, total4);

    k_wt<<<(3 * DD * DD) / 256, 256, 0, stream>>>(Wq_w, Wk_w, Wo_w, WqT, WkT, WoT);

    int mblocks = (N + 15) / 16;
    dim3 gproj(mblocks, 2);
    k_proj<<<gproj, 256, 0, stream>>>(h16, WqT, WkT, Wq_b, Wk_b, q16, k16, N);

    k_attn<<<(N + 7) / 8, 256, 0, stream>>>(h, nbr_idx, nbr_mask, q16, k16, Wk_b, ctx16, N);

    k_out<<<mblocks, 256, 0, stream>>>(ctx16, WoT, Wo_b, gamma, beta, out, N);
}
